// TemEnc_5514738008902
// MI455X (gfx1250) — matrix operand order repaired and checked by reading, not compiled, not run on hardware
//
#include <hip/hip_runtime.h>
#include <hip/hip_bf16.h>
#include <math.h>

typedef __attribute__((ext_vector_type(16))) _Float16 v16h;
typedef __attribute__((ext_vector_type(8)))  _Float16 v8h;
typedef __attribute__((ext_vector_type(4)))  _Float16 v4h;
typedef __attribute__((ext_vector_type(2)))  _Float16 v2h;
typedef __attribute__((ext_vector_type(8)))  float    v8f;
typedef __attribute__((ext_vector_type(4)))  int      v4i;

#define B_    64
#define L_    4096
#define C_    128
#define WIN   64
#define NKEEP 2048   // L - tr_count

#if defined(__HIP_DEVICE_COMPILE__) && __has_builtin(__builtin_amdgcn_global_load_async_to_lds_b128)
#define HAVE_ASYNC_LDS 1
#else
#define HAVE_ASYNC_LDS 0
#endif

#if HAVE_ASYNC_LDS
typedef __attribute__((address_space(1))) v4i g1v4i;   // global 16B chunk
typedef __attribute__((address_space(3))) v4i l3v4i;   // LDS 16B chunk
// Per-lane async copy of 16B global -> LDS, tracked by ASYNCcnt (no VGPR round trip).
__device__ __forceinline__ void async_g2l_b128(const void* g, void* l) {
    __builtin_amdgcn_global_load_async_to_lds_b128(
        (g1v4i*)(uintptr_t)g,
        (l3v4i*)(uint32_t)(uintptr_t)l,   // low 32 bits of generic LDS addr = LDS offset
        0, 0);
}
__device__ __forceinline__ void wait_async0() {
#if __has_builtin(__builtin_amdgcn_s_wait_asynccnt)
    __builtin_amdgcn_s_wait_asynccnt(0);
#else
    asm volatile("s_wait_asynccnt 0" ::: "memory");
#endif
}
#else
__device__ __forceinline__ void wait_async0() {}
#endif

__device__ __forceinline__ float gelu_exact(float x) {
    return 0.5f * x * (1.0f + erff(x * 0.70710678118654752f));
}
__device__ __forceinline__ float sigmoid_(float x) {
    return 1.0f / (1.0f + __expf(-x));
}
__device__ __forceinline__ v8f zero8() {
    v8f z;
#pragma unroll
    for (int i = 0; i < 8; ++i) z[i] = 0.0f;
    return z;
}

// A fragment (16x32 f16): lane = m + 16h holds contiguous K = k0+16h .. +15 of row m.
__device__ __forceinline__ v16h load_a_frag(const _Float16* base, int row, int stride,
                                            int k0, int h) {
    return *(const v16h*)(base + row * stride + k0 + h * 16);
}
// B fragment (32x16 f16): lane = n + 16h holds contiguous K = k0+16h .. +15 of column n.
__device__ __forceinline__ v16h load_b_frag(const _Float16* base, int col, int stride,
                                            int k0, int h) {
    return *(const v16h*)(base + col * stride + k0 + h * 16);
}
__device__ __forceinline__ v8f wmma_f16(v16h a, v16h b, v8f c) {
    return __builtin_amdgcn_wmma_f32_16x16x32_f16(false, a, false, b, (short)0, c,
                                                  false, false);
}

// ---------------- score: sliding-window mean/var, score = sum_c v / sum_c m ----------
__global__ __launch_bounds__(128) void score_kernel(const float* __restrict__ x,
                                                    float* __restrict__ score) {
    const int b = blockIdx.x, c = threadIdx.x;
    const int lane = c & 31, wave = c >> 5;
    const float* xb = x + (size_t)b * L_ * C_;
    __shared__ float red[4][3];
    float S = 0.f, S2 = 0.f;
    for (int t = 0; t < L_; ++t) {
        float v = xb[(size_t)t * C_ + c];
        S += v; S2 += v * v;
        if (t >= WIN) {
            float o = xb[(size_t)(t - WIN) * C_ + c];
            S -= o; S2 -= o * o;
        }
        float a0 = S, a1 = S2, a2 = S * S;
#pragma unroll
        for (int off = 16; off >= 1; off >>= 1) {
            a0 += __shfl_xor(a0, off, 32);
            a1 += __shfl_xor(a1, off, 32);
            a2 += __shfl_xor(a2, off, 32);
        }
        if (lane == 0) { red[wave][0] = a0; red[wave][1] = a1; red[wave][2] = a2; }
        __syncthreads();
        if (c == 0) {
            float s0 = red[0][0] + red[1][0] + red[2][0] + red[3][0];
            float s1 = red[0][1] + red[1][1] + red[2][1] + red[3][1];
            float s2 = red[0][2] + red[1][2] + red[2][2] + red[3][2];
            float cnt = (float)((t + 1 < WIN) ? (t + 1) : WIN);
            score[(size_t)b * L_ + t] = (s1 - s2 / cnt) / s0;
        }
        __syncthreads();
    }
}

// ---------------- rank select: deterministic top_k(-score) ordering ------------------
__global__ __launch_bounds__(256) void rank_kernel(const float* __restrict__ score,
                                                   int* __restrict__ idx) {
    __shared__ float s[L_];
    const int b = blockIdx.x;
    for (int i = threadIdx.x; i < L_; i += 256) s[i] = score[(size_t)b * L_ + i];
    __syncthreads();
    for (int i = threadIdx.x; i < L_; i += 256) {
        float si = s[i];
        int r = 0;
        for (int j = 0; j < L_; ++j) {
            float sj = s[j];
            r += (sj < si) || (sj == si && j < i);   // ascending score, ties by index
        }
        if (r < NKEEP) idx[(size_t)b * NKEEP + r] = i;
    }
}

// ---------------- gather selected rows -> f16 ---------------------------------------
__global__ __launch_bounds__(256) void gather_kernel(const float* __restrict__ x,
                                                     const int* __restrict__ idx,
                                                     _Float16* __restrict__ g) {
    size_t e = (size_t)blockIdx.x * 256 + threadIdx.x;   // B*NKEEP*C elements
    int c = (int)(e & (C_ - 1));
    size_t r = e >> 7;
    int i = (int)(r & (NKEEP - 1));
    int b = (int)(r >> 11);
    int src = idx[(size_t)b * NKEEP + i];
    g[e] = (_Float16)x[((size_t)b * L_ + src) * C_ + c];
}

// ---------------- weight transpose + convert: Wt[n][k] = W[k][n] --------------------
__global__ __launch_bounds__(256) void wconv_kernel(const float* __restrict__ W,
                                                    _Float16* __restrict__ Wt) {
    int e = blockIdx.x * 256 + threadIdx.x;   // 16384
    int n = e >> 7, k = e & 127;
    Wt[n * 128 + k] = (_Float16)W[k * 128 + n];
}

// ---------------- fused GEMM: out = act(A[M,128] @ W[128,128] + bias) ---------------
// Block = 256 thr (8 waves), block tile M=128, N=128; wave w owns rows w*16..+15.
template <int ACT, bool A_IS_F32, bool OUT_F16>
__global__ __launch_bounds__(256) void gemm128_kernel(const void* __restrict__ Ap,
                                                      const _Float16* __restrict__ Wt,
                                                      const float* __restrict__ bias,
                                                      void* __restrict__ outp) {
    __shared__ _Float16 sW[128 * 128];   // N-major weights, 32 KB
    __shared__ _Float16 sA[128 * 128];   // row-major A tile, 32 KB
    const int t = threadIdx.x;
    const size_t row0 = (size_t)blockIdx.x * 128;

    // ---- weight tile -> LDS ----
#if HAVE_ASYNC_LDS
#pragma unroll
    for (int i = 0; i < 8; ++i) {
        int id = (t + i * 256) * 8;                 // 8 f16 = 16B per op
        async_g2l_b128(Wt + id, sW + id);
    }
#else
#pragma unroll
    for (int i = 0; i < 8; ++i)
        ((uint4*)sW)[t + i * 256] = ((const uint4*)Wt)[t + i * 256];
#endif

    // ---- A tile -> LDS ----
    if (A_IS_F32) {
        const float* A = (const float*)Ap;
#pragma unroll
        for (int i = 0; i < 16; ++i) {
            int id = t + i * 256;
            int m = id >> 5;
            int k = (id & 31) * 4;
            float4 v = *(const float4*)(A + (row0 + m) * 128 + k);
            v4h p;
            p[0] = (_Float16)v.x; p[1] = (_Float16)v.y;
            p[2] = (_Float16)v.z; p[3] = (_Float16)v.w;
            *(v4h*)(sA + m * 128 + k) = p;          // one packed b64 DS store
        }
    } else {
        const _Float16* A = (const _Float16*)Ap + row0 * 128;
#if HAVE_ASYNC_LDS
#pragma unroll
        for (int i = 0; i < 8; ++i) {
            int id = (t + i * 256) * 8;
            async_g2l_b128(A + id, sA + id);
        }
#else
#pragma unroll
        for (int i = 0; i < 8; ++i)
            ((uint4*)sA)[t + i * 256] = ((const uint4*)A)[t + i * 256];
#endif
    }
    wait_async0();
    __syncthreads();

    const int wave = t >> 5, lane = t & 31;
    const int m16 = lane & 15, h = lane >> 4;
    const int mrow = wave * 16;
    v8f acc[8];
#pragma unroll
    for (int nt = 0; nt < 8; ++nt) acc[nt] = zero8();
#pragma unroll
    for (int k0 = 0; k0 < 128; k0 += 32) {
        v16h a = load_a_frag(sA, mrow + m16, 128, k0, h);
#pragma unroll
        for (int nt = 0; nt < 8; ++nt) {
            v16h bf = load_b_frag(sW, nt * 16 + m16, 128, k0, h);
            acc[nt] = wmma_f16(a, bf, acc[nt]);
        }
    }
#pragma unroll
    for (int nt = 0; nt < 8; ++nt) {
        int col = nt * 16 + m16;
        float bv = bias[col];
#pragma unroll
        for (int r = 0; r < 8; ++r) {
            size_t row = row0 + mrow + r + 8 * h;
            float v = acc[nt][r] + bv;
            v = (ACT == 0) ? gelu_exact(v) : sigmoid_(v);
            if (OUT_F16) ((_Float16*)outp)[row * 128 + col] = (_Float16)v;
            else         ((float*)outp)[row * 128 + col] = v;
        }
    }
}

// ---------------- causal graph: G = (X^T X) / 64 per batch (pre-softmax) ------------
// One block per batch; waves tiled 4(M)x2(N); both A and B frags come from the
// transposed LDS tile sXT[c][l] (K = l is contiguous for both).
__global__ __launch_bounds__(256) void cg_gemm_kernel(const float* __restrict__ x,
                                                      float* __restrict__ G) {
    __shared__ _Float16 sXT[128 * 32];   // [c][l], 8 KB
    const int b = blockIdx.x, t = threadIdx.x;
    const int wave = t >> 5, lane = t & 31, m16 = lane & 15, h = lane >> 4;
    const int mg = wave >> 1, ng = wave & 1;
    const float* xb = x + (size_t)b * L_ * C_;
    v8f acc[2][4];
#pragma unroll
    for (int mi = 0; mi < 2; ++mi)
#pragma unroll
        for (int ni = 0; ni < 4; ++ni) acc[mi][ni] = zero8();

    for (int k0 = 0; k0 < L_; k0 += 32) {
        __syncthreads();
        // 2048 (c, l-pair) items over 256 threads: coalesced global reads of two
        // consecutive rows, one packed b32 LDS store per pair.
#pragma unroll
        for (int i = 0; i < 8; ++i) {
            int e = t + i * 256;          // 0..2047
            int c = e & 127;
            int l = (e >> 7) * 2;         // 0,2,..,30
            float a = xb[(size_t)(k0 + l) * 128 + c];
            float bb = xb[(size_t)(k0 + l + 1) * 128 + c];
            v2h p; p[0] = (_Float16)a; p[1] = (_Float16)bb;
            *(v2h*)(sXT + c * 32 + l) = p;
        }
        if (k0 + 32 < L_)                  // prefetch next K tile (global_prefetch_b8)
            __builtin_prefetch(xb + (size_t)(k0 + 32) * 128 + t * 16, 0, 0);
        __syncthreads();
#pragma unroll
        for (int mi = 0; mi < 2; ++mi) {
            v16h a = load_a_frag(sXT, mg * 32 + mi * 16 + m16, 32, 0, h);
#pragma unroll
            for (int ni = 0; ni < 4; ++ni) {
                v16h bf = load_b_frag(sXT, ng * 64 + ni * 16 + m16, 32, 0, h);
                acc[mi][ni] = wmma_f16(a, bf, acc[mi][ni]);
            }
        }
    }
    float* Gb = G + (size_t)b * 128 * 128;
#pragma unroll
    for (int mi = 0; mi < 2; ++mi)
#pragma unroll
        for (int ni = 0; ni < 4; ++ni)
#pragma unroll
            for (int r = 0; r < 8; ++r) {
                int row = mg * 32 + mi * 16 + r + 8 * h;
                int col = ng * 64 + ni * 16 + m16;
                Gb[row * 128 + col] = acc[mi][ni][r] * 0.015625f;  // 1/sqrt(4096)
            }
}

__global__ __launch_bounds__(128) void cg_softmax_kernel(const float* __restrict__ G,
                                                         float* __restrict__ out) {
    const int b = blockIdx.x, r = threadIdx.x;   // one row per thread
    const float* row = G + ((size_t)b * 128 + r) * 128;
    float mx = -1e30f;
    for (int j = 0; j < 128; ++j) mx = fmaxf(mx, row[j]);
    float sum = 0.f;
    for (int j = 0; j < 128; ++j) sum += __expf(row[j] - mx);
    float inv = 1.0f / sum;
    float* orow = out + ((size_t)b * 128 + r) * 128;
    for (int j = 0; j < 128; ++j) orow[j] = __expf(row[j] - mx) * inv;
}

extern "C" void kernel_launch(void* const* d_in, const int* in_sizes, int n_in,
                              void* d_out, int out_size, void* d_ws, size_t ws_size,
                              hipStream_t stream) {
    const float* x  = (const float*)d_in[0];
    const float* We = (const float*)d_in[1];
    const float* be = (const float*)d_in[2];
    const float* W1 = (const float*)d_in[3];
    const float* b1 = (const float*)d_in[4];
    const float* W2 = (const float*)d_in[5];
    const float* b2 = (const float*)d_in[6];

    char* ws = (char*)d_ws;
    float*    score = (float*)(ws);                               // 1,048,576 B
    int*      idx   = (int*)(ws + 1048576);                       //   524,288 B
    _Float16* WeT   = (_Float16*)(ws + 1572864);                  //    32,768 B
    _Float16* W1T   = (_Float16*)(ws + 1605632);                  //    32,768 B
    _Float16* W2T   = (_Float16*)(ws + 1638400);                  //    32,768 B
    _Float16* g16   = (_Float16*)(ws + 1671168);                  // 33,554,432 B
    _Float16* u16   = (_Float16*)(ws + 1671168 + 33554432);       // 33,554,432 B
    float*    Gtmp  = (float*)(ws + 1671168 + 2 * 33554432);      //  4,194,304 B

    float* rec    = (float*)d_out;                                // [64,2048,128]
    float* output = rec + (size_t)B_ * NKEEP * C_;                // [64,4096,128]
    float* cg     = output + (size_t)B_ * L_ * C_;                // [64,128,128]

    wconv_kernel<<<64, 256, 0, stream>>>(We, WeT);
    wconv_kernel<<<64, 256, 0, stream>>>(W1, W1T);
    wconv_kernel<<<64, 256, 0, stream>>>(W2, W2T);

    score_kernel<<<B_, 128, 0, stream>>>(x, score);
    rank_kernel<<<B_, 256, 0, stream>>>(score, idx);
    gather_kernel<<<(B_ * NKEEP * C_) / 256, 256, 0, stream>>>(x, idx, g16);

    // output = gelu(x @ We + be)
    gemm128_kernel<0, true, false><<<(B_ * L_) / 128, 256, 0, stream>>>(x, WeT, be, output);
    // ux = gelu(g @ We + be)
    gemm128_kernel<0, false, true><<<(B_ * NKEEP) / 128, 256, 0, stream>>>(g16, WeT, be, u16);
    // h1 = gelu(ux @ W1 + b1)  (reuse g16 buffer)
    gemm128_kernel<0, false, true><<<(B_ * NKEEP) / 128, 256, 0, stream>>>(u16, W1T, b1, g16);
    // rec = sigmoid(h1 @ W2 + b2)
    gemm128_kernel<1, false, false><<<(B_ * NKEEP) / 128, 256, 0, stream>>>(g16, W2T, b2, rec);

    cg_gemm_kernel<<<B_, 256, 0, stream>>>(x, Gtmp);
    cg_softmax_kernel<<<B_, 128, 0, stream>>>(Gtmp, cg);
}
